// AutoregressiveSubsetSampler_71519795413539
// MI455X (gfx1250) — matrix operand order repaired and checked by reading, not compiled, not run on hardware
//
#include <hip/hip_runtime.h>

// ---------------------------------------------------------------------------
// Types for CDNA5 WMMA
// ---------------------------------------------------------------------------
typedef __attribute__((ext_vector_type(16))) __bf16        v16bf;
typedef __attribute__((ext_vector_type(8)))  float         v8f;
typedef __attribute__((ext_vector_type(8)))  unsigned int  v8u;

#define T_STEPS   8192   // FACTOR * N_NODES
#define N_NODES_C 4096
#define SIZE_G    256
#define N_ROW     2048
#define D_FF      2048
#define NT1       1024   // phase-1 block size

// ---------------------------------------------------------------------------
// Helpers
// ---------------------------------------------------------------------------
__device__ __forceinline__ float bflo(unsigned int q) {
  return __builtin_bit_cast(float, q << 16);
}
__device__ __forceinline__ float bfhi(unsigned int q) {
  return __builtin_bit_cast(float, q & 0xFFFF0000u);
}
__device__ __forceinline__ unsigned short f2bf(float f) {
  unsigned int u = __builtin_bit_cast(unsigned int, f);
  u = (u + 0x7FFFu + ((u >> 16) & 1u)) >> 16;   // round-to-nearest-even
  return (unsigned short)u;
}

// ---------------------------------------------------------------------------
// Prep kernels: fp32 -> bf16 conversion and on-device weight fusion
// ---------------------------------------------------------------------------
__global__ void cvt_bf16(const float* __restrict__ src,
                         unsigned short* __restrict__ dst, int n) {
  for (int i = blockIdx.x * blockDim.x + threadIdx.x; i < n;
       i += gridDim.x * blockDim.x)
    dst[i] = f2bf(src[i]);
}

// C[M,N] (bf16) = A[M,K] @ B[K,N], all fp32 row-major inputs
__global__ void fuse_mm(const float* __restrict__ A, const float* __restrict__ B,
                        unsigned short* __restrict__ C, int M, int K, int N) {
  int idx = blockIdx.x * blockDim.x + threadIdx.x;
  if (idx >= M * N) return;
  int i = idx / N, j = idx % N;
  float acc = 0.f;
  for (int k = 0; k < K; ++k) acc += A[i * K + k] * B[k * N + j];
  C[idx] = f2bf(acc);
}

// bg2 = Wg@b2 + bg ; bov = Wo@bv + bo   (256 threads, 1 block)
__global__ void fuse_bias(const float* __restrict__ Wg, const float* __restrict__ b2,
                          const float* __restrict__ bg, const float* __restrict__ Wo,
                          const float* __restrict__ bv, const float* __restrict__ bo,
                          float* __restrict__ bg2, float* __restrict__ bov) {
  int i = threadIdx.x;
  float s1 = bg[i], s2 = bo[i];
  for (int k = 0; k < 256; ++k) {
    s1 += Wg[i * 256 + k] * b2[k];
    s2 += Wo[i * 256 + k] * bv[k];
  }
  bg2[i] = s1;
  bov[i] = s2;
}

// ---------------------------------------------------------------------------
// Phase 1: sequential scan (single workgroup, 1024 threads, LDS activations)
// ---------------------------------------------------------------------------
__device__ float block_sum(float v, float* red) {
  const int tid = threadIdx.x;
  red[tid] = v;
  __syncthreads();
  for (int s = 512; s > 0; s >>= 1) {
    if (tid < s) red[tid] += red[tid + s];
    __syncthreads();
  }
  float r = red[0];
  __syncthreads();
  return r;
}

// y[out] = act(W_bf16[out,in] @ x + bias); all 1024 threads participate.
__device__ void matvec_bf16(const unsigned short* __restrict__ W,
                            const float* __restrict__ bias,
                            const float* __restrict__ x, float* __restrict__ y,
                            float* __restrict__ part, int out, int in,
                            bool do_relu) {
  const int tid = threadIdx.x;
  if (out >= NT1) {                       // e.g. Wf1: 2048 rows
    const int rpt = out / NT1;
    for (int rr = 0; rr < rpt; ++rr) {
      const int row = tid + rr * NT1;
      float acc = 0.f;
      const uint4* p = reinterpret_cast<const uint4*>(W + (size_t)row * in);
      for (int k = 0; k < in; k += 8) {
        uint4 q = p[k >> 3];
        const float* xs = x + k;
        acc += bflo(q.x) * xs[0]; acc += bfhi(q.x) * xs[1];
        acc += bflo(q.y) * xs[2]; acc += bfhi(q.y) * xs[3];
        acc += bflo(q.z) * xs[4]; acc += bfhi(q.z) * xs[5];
        acc += bflo(q.w) * xs[6]; acc += bfhi(q.w) * xs[7];
      }
      acc += bias[row];
      if (do_relu) acc = fmaxf(acc, 0.f);
      y[row] = acc;
    }
    __syncthreads();
  } else {                                // split-K across spare lanes
    const int split = NT1 / out;          // 2 (out=512) or 4 (out=256)
    const int row = tid % out;
    const int seg = tid / out;
    const int len = in / split;
    const int k0  = seg * len;
    float acc = 0.f;
    const uint4* p = reinterpret_cast<const uint4*>(W + (size_t)row * in + k0);
    for (int k = 0; k < len; k += 8) {
      uint4 q = p[k >> 3];
      const float* xs = x + k0 + k;
      acc += bflo(q.x) * xs[0]; acc += bfhi(q.x) * xs[1];
      acc += bflo(q.y) * xs[2]; acc += bfhi(q.y) * xs[3];
      acc += bflo(q.z) * xs[4]; acc += bfhi(q.z) * xs[5];
      acc += bflo(q.w) * xs[6]; acc += bfhi(q.w) * xs[7];
    }
    part[tid] = acc;
    __syncthreads();
    if (tid < out) {
      float s = 0.f;
      for (int sg = 0; sg < split; ++sg) s += part[sg * out + tid];
      s += bias[tid];
      if (do_relu) s = fmaxf(s, 0.f);
      y[tid] = s;
    }
    __syncthreads();
  }
}

__device__ void layernorm256(const float* __restrict__ x, float* __restrict__ y,
                             const float* __restrict__ gw,
                             const float* __restrict__ bw, float* red) {
  const int tid = threadIdx.x;
  float v = (tid < 256) ? x[tid] : 0.f;
  float mu = block_sum(v, red) * (1.f / 256.f);
  float d = (tid < 256) ? (x[tid] - mu) : 0.f;
  float var = block_sum(d * d, red) * (1.f / 256.f);
  float rstd = rsqrtf(var + 1e-5f);
  if (tid < 256) y[tid] = (x[tid] - mu) * rstd * gw[tid] + bw[tid];
  __syncthreads();
}

__global__ __launch_bounds__(NT1) void phase1_scan(
    const float* __restrict__ m,           // gnn_embeds [4096,256]
    const unsigned short* __restrict__ W1b, const float* __restrict__ b1,
    const unsigned short* __restrict__ W2b, const float* __restrict__ b2,
    const unsigned short* __restrict__ Wg2b, const float* __restrict__ bg2,
    const unsigned short* __restrict__ Wovb, const float* __restrict__ bov,
    const unsigned short* __restrict__ Wf1b, const float* __restrict__ bf1,
    const unsigned short* __restrict__ Wf2b, const float* __restrict__ bf2,
    const float* __restrict__ ln1g, const float* __restrict__ ln1b,
    const float* __restrict__ ln2g, const float* __restrict__ ln2b,
    unsigned short* __restrict__ H)        // out: [8192,256] bf16
{
  __shared__ float x512[512], h1[512], hbuf[256], gn[256], sa[256];
  __shared__ float xln[256], f1[2048], ff[256], g[256], tmp[256];
  __shared__ float part[NT1], red[NT1];
  const int tid = threadIdx.x;

  if (tid < 256) g[tid] = 0.f;            // g0 = zeros
  __syncthreads();

  for (int t = 0; t < T_STEPS; ++t) {
    if (tid < 256) {
      x512[tid]       = m[(size_t)(t & (N_NODES_C - 1)) * 256 + tid];
      x512[256 + tid] = g[tid];
    }
    __syncthreads();
    // h1 = relu(W1 @ [m;g] + b1)
    matvec_bf16(W1b, b1, x512, h1, part, 512, 512, true);
    // h = W2 @ h1 + b2  (only needed for deferred sampling head)
    matvec_bf16(W2b, b2, h1, hbuf, part, 256, 512, false);
    if (tid < 256) H[(size_t)t * 256 + tid] = f2bf(hbuf[tid]);
    // g_new = Wg2 @ h1 + bg2   (Wg2 = Wg @ W2 fused)
    matvec_bf16(Wg2b, bg2, h1, gn, part, 256, 512, false);
    // sa = Wov @ g_new + bov   (Wov = Wo @ Wv fused; seq_len==1 attention)
    matvec_bf16(Wovb, bov, gn, sa, part, 256, 256, false);
    if (tid < 256) tmp[tid] = gn[tid] + sa[tid];
    __syncthreads();
    layernorm256(tmp, xln, ln1g, ln1b, red);
    // ff = Wf2 @ relu(Wf1 @ x + bf1) + bf2
    matvec_bf16(Wf1b, bf1, xln, f1, part, 2048, 256, true);
    matvec_bf16(Wf2b, bf2, f1, ff, part, 256, 2048, false);
    if (tid < 256) tmp[tid] = xln[tid] + ff[tid];
    __syncthreads();
    layernorm256(tmp, g, ln2g, ln2b, red);
  }
}

// ---------------------------------------------------------------------------
// Phase 2a: probs = sigmoid(H @ Wr^T + br) via WMMA bf16, staged into d_out
// ---------------------------------------------------------------------------
// A fragment: 16x32 bf16, lanes 0-15 -> M, K pairs {0..7,16..23};
//             lanes 16-31 -> same M set, K pairs {8..15,24..31}
__device__ __forceinline__ v16bf load_a_frag(const unsigned short* __restrict__ H,
                                             int t0, int kbase, int lane) {
  const int mrow  = lane & 15;
  const int khalf = (lane < 16) ? 0 : 8;
  const unsigned short* p = H + (size_t)(t0 + mrow) * 256 + kbase + khalf;
  v8u u;
  u[0] = *(const unsigned int*)(p + 0);
  u[1] = *(const unsigned int*)(p + 2);
  u[2] = *(const unsigned int*)(p + 4);
  u[3] = *(const unsigned int*)(p + 6);
  u[4] = *(const unsigned int*)(p + 16);
  u[5] = *(const unsigned int*)(p + 18);
  u[6] = *(const unsigned int*)(p + 20);
  u[7] = *(const unsigned int*)(p + 22);
  return __builtin_bit_cast(v16bf, u);
}

// B fragment: 32x16 bf16 (K x N); B[k,n] = Wr[n,k] so Wr row-major works.
// lanes 0-15 -> N=lane, K={0..7,16..23} ; lanes 16-31 -> N=lane-16, K={8..15,24..31}
__device__ __forceinline__ v16bf load_b_frag(const unsigned short* __restrict__ Wr,
                                             int r0, int kbase, int lane) {
  const int n     = lane & 15;
  const int khalf = (lane < 16) ? 0 : 8;
  const unsigned short* p = Wr + (size_t)(r0 + n) * 256 + kbase + khalf;
  v8u u;
#pragma unroll
  for (int v = 0; v < 8; ++v) u[v] = *(const unsigned int*)(p + 2 * v + (v >= 4 ? 8 : 0));
  return __builtin_bit_cast(v16bf, u);
}

__global__ __launch_bounds__(256) void phase2a_gemm(
    const unsigned short* __restrict__ H, const unsigned short* __restrict__ Wrb,
    const float* __restrict__ br, float* __restrict__ probs_out) {
  const int tid  = threadIdx.x;
  const int lane = tid & 31;
  const int w    = tid >> 5;                 // wave id 0..7
  const int t0   = blockIdx.x * 16;          // 512 t-tiles
  const int r0   = (blockIdx.y * 8 + w) * 16;  // 16 r-groups x 8 waves = 128 tiles

  __builtin_prefetch(Wrb + (size_t)r0 * 256, 0, 1);

  v16bf afrag[8];
#pragma unroll
  for (int c = 0; c < 8; ++c) afrag[c] = load_a_frag(H, t0, c * 32, lane);

  v8f acc = {0.f, 0.f, 0.f, 0.f, 0.f, 0.f, 0.f, 0.f};
#pragma unroll
  for (int c = 0; c < 8; ++c) {
    v16bf b = load_b_frag(Wrb, r0, c * 32, lane);
    acc = __builtin_amdgcn_wmma_f32_16x16x32_bf16(
        /*neg_a=*/false, afrag[c], /*neg_b=*/false, b,
        /*c_mod=*/(short)0, acc, /*reuse_a=*/false, /*reuse_b=*/false);
  }

  const int col   = r0 + (lane & 15);
  const int mbase = (lane < 16) ? 0 : 8;
  const float bias = br[col];
#pragma unroll
  for (int v = 0; v < 8; ++v) {
    float logit = acc[v] + bias;
    probs_out[(size_t)(t0 + mbase + v) * N_ROW + col] =
        1.f / (1.f + __expf(-logit));
  }
}

// ---------------------------------------------------------------------------
// Phase 2b: per-row rejection sampling; overwrites probs row with 0/1 row
// ---------------------------------------------------------------------------
__global__ __launch_bounds__(256) void phase2b_sample(
    const float* __restrict__ u, float* __restrict__ inout) {
  __shared__ float prow[N_ROW];
  __shared__ int cnt[4];
  __shared__ int sel_s;
  const int t = blockIdx.x;
  const int tid = threadIdx.x;
  float* row = inout + (size_t)t * N_ROW;

  for (int r = tid; r < N_ROW; r += 256) prow[r] = row[r];
  if (tid < 4) cnt[tid] = 0;
  __syncthreads();

  const float* ut = u + (size_t)t * 4 * N_ROW;
  int pc0 = 0, pc1 = 0, pc2 = 0, pc3 = 0;
  for (int r = tid; r < N_ROW; r += 256) {
    float p = prow[r];
    pc0 += (ut[r] < p);
    pc1 += (ut[N_ROW + r] < p);
    pc2 += (ut[2 * N_ROW + r] < p);
    pc3 += (ut[3 * N_ROW + r] < p);
  }
  atomicAdd(&cnt[0], pc0);
  atomicAdd(&cnt[1], pc1);
  atomicAdd(&cnt[2], pc2);
  atomicAdd(&cnt[3], pc3);
  __syncthreads();

  if (tid == 0) {
    int sel = 0;                            // argmax over all-false -> 0
    for (int a = 0; a < 4; ++a) {
      int c = cnt[a];
      if ((c == 0) || (c >= 2 && c <= 6)) { sel = a; break; }
    }
    sel_s = sel;
  }
  __syncthreads();

  const float* us = ut + (size_t)sel_s * N_ROW;
  for (int r = tid; r < N_ROW; r += 256)
    row[r] = (us[r] < prow[r]) ? 1.f : 0.f;
}

// ---------------------------------------------------------------------------
// Host launcher
// ---------------------------------------------------------------------------
extern "C" void kernel_launch(void* const* d_in, const int* in_sizes, int n_in,
                              void* d_out, int out_size, void* d_ws,
                              size_t ws_size, hipStream_t stream) {
  (void)in_sizes; (void)n_in; (void)out_size; (void)ws_size;

  const float* gnn  = (const float*)d_in[0];
  const float* u    = (const float*)d_in[2];
  const float* W1   = (const float*)d_in[3];
  const float* b1   = (const float*)d_in[4];
  const float* W2   = (const float*)d_in[5];
  const float* b2   = (const float*)d_in[6];
  const float* Wr   = (const float*)d_in[7];
  const float* br   = (const float*)d_in[8];
  const float* Wg   = (const float*)d_in[9];
  const float* bg   = (const float*)d_in[10];
  const float* Wv   = (const float*)d_in[11];
  const float* bv   = (const float*)d_in[12];
  const float* Wo   = (const float*)d_in[13];
  const float* bo   = (const float*)d_in[14];
  const float* Wf1  = (const float*)d_in[15];
  const float* bf1  = (const float*)d_in[16];
  const float* Wf2  = (const float*)d_in[17];
  const float* bf2  = (const float*)d_in[18];
  const float* ln1g = (const float*)d_in[19];
  const float* ln1b = (const float*)d_in[20];
  const float* ln2g = (const float*)d_in[21];
  const float* ln2b = (const float*)d_in[22];
  float* out = (float*)d_out;

  // Workspace carve-up (256B aligned), ~8.5 MB total
  char* ws = (char*)d_ws;
  size_t off = 0;
  auto take = [&](size_t bytes) -> char* {
    char* p = ws + off;
    off += (bytes + 255) & ~(size_t)255;
    return p;
  };
  unsigned short* Hb   = (unsigned short*)take((size_t)T_STEPS * 256 * 2);
  unsigned short* W1b  = (unsigned short*)take(512 * 512 * 2);
  unsigned short* W2b  = (unsigned short*)take(256 * 512 * 2);
  unsigned short* Wg2b = (unsigned short*)take(256 * 512 * 2);
  unsigned short* Wovb = (unsigned short*)take(256 * 256 * 2);
  unsigned short* Wf1b = (unsigned short*)take(2048 * 256 * 2);
  unsigned short* Wf2b = (unsigned short*)take(256 * 2048 * 2);
  unsigned short* Wrb  = (unsigned short*)take(2048 * 256 * 2);
  float* bg2 = (float*)take(256 * 4);
  float* bov = (float*)take(256 * 4);

  // --- Prep: bf16 conversions + fused weights ---
  auto cvt = [&](const float* s, unsigned short* d, int n) {
    cvt_bf16<<<(n + 255) / 256, 256, 0, stream>>>(s, d, n);
  };
  cvt(W1,  W1b,  512 * 512);
  cvt(W2,  W2b,  256 * 512);
  cvt(Wf1, Wf1b, 2048 * 256);
  cvt(Wf2, Wf2b, 256 * 2048);
  cvt(Wr,  Wrb,  2048 * 256);
  // Wg2 = Wg @ W2  [256,512]
  fuse_mm<<<(256 * 512 + 255) / 256, 256, 0, stream>>>(Wg, W2, Wg2b, 256, 256, 512);
  // Wov = Wo @ Wv  [256,256]
  fuse_mm<<<(256 * 256 + 255) / 256, 256, 0, stream>>>(Wo, Wv, Wovb, 256, 256, 256);
  fuse_bias<<<1, 256, 0, stream>>>(Wg, b2, bg, Wo, bv, bo, bg2, bov);

  // --- Phase 1: sequential scan, single workgroup ---
  phase1_scan<<<1, NT1, 0, stream>>>(gnn, W1b, b1, W2b, b2, Wg2b, bg2, Wovb,
                                     bov, Wf1b, bf1, Wf2b, bf2, ln1g, ln1b,
                                     ln2g, ln2b, Hb);

  // --- Phase 2a: WMMA GEMM -> probs staged in d_out ---
  dim3 g2a(T_STEPS / 16, N_ROW / (16 * 8));   // (512, 16)
  phase2a_gemm<<<g2a, 256, 0, stream>>>(Hb, Wrb, br, out);

  // --- Phase 2b: rejection sampling, overwrite d_out with incidence rows ---
  phase2b_sample<<<T_STEPS, 256, 0, stream>>>(u, out);
}
